// BalancedSpikingCell_18176301596897
// MI455X (gfx1250) — matrix operand order repaired and checked by reading, not compiled, not run on hardware
//
#include <hip/hip_runtime.h>

typedef _Float16 f16;
typedef _Float16 v4h  __attribute__((ext_vector_type(4)));
typedef _Float16 v8h  __attribute__((ext_vector_type(8)));
typedef _Float16 v16h __attribute__((ext_vector_type(16)));
typedef float    v4f  __attribute__((ext_vector_type(4)));
typedef float    v8f  __attribute__((ext_vector_type(8)));
typedef int      b128i __attribute__((vector_size(16)));   // matches builtin proto

#define B_DIM 2048
#define NE    4096
#define NI    1024
#define DIN   1024
#define KTOT  6144   // DIN + NE + NI
#define NTOT  5120   // NE + NI

// d_out flat layout (floats), in reference return order
#define OFF_SE 0L
#define OFF_SI ((long)B_DIM * NE)                 // 8388608
#define OFF_VE (OFF_SI + (long)B_DIM * NI)        // 10485760
#define OFF_IE (OFF_VE + (long)B_DIM * NE)        // 18874368
#define OFF_VI (OFF_IE + (long)B_DIM * NE)        // 27262976
#define OFF_II (OFF_VI + (long)B_DIM * NI)        // 29360128

// LDS tile row strides (f16 elements): 32 data + 8 pad -> 80B row stride,
// keeps 16B alignment for b128 and spreads the 16 row-lanes across banks.
#define LDA 40
#define LDB 40
#define NSTAGES (KTOT / 32)   // 192

// ---------------------------------------------------------------------------
// CDNA5 async global->LDS copy (ASYNCcnt-tracked), with sync fallback.
// ---------------------------------------------------------------------------
#if defined(__has_builtin)
#if __has_builtin(__builtin_amdgcn_global_load_async_to_lds_b128) && \
    __has_builtin(__builtin_amdgcn_s_wait_asynccnt)
#define HAS_ASYNC_LDS 1
#endif
#endif

__device__ __forceinline__ void cp16_g2lds(f16* dst_lds, const f16* src_g) {
#ifdef HAS_ASYNC_LDS
    __builtin_amdgcn_global_load_async_to_lds_b128(
        (__attribute__((address_space(1))) b128i*)(f16*)src_g,
        (__attribute__((address_space(3))) b128i*)dst_lds,
        0, 0);
#else
    *(v8h*)dst_lds = *(const v8h*)src_g;
#endif
}

#ifdef HAS_ASYNC_LDS
#define ASYNC_WAIT(n) __builtin_amdgcn_s_wait_asynccnt(n)
#else
#define ASYNC_WAIT(n) do {} while (0)
#endif

// ---------------------------------------------------------------------------
// Pack A = [x, spikes_e, -spikes_i] as f16, row-major [B_DIM, KTOT]
// ---------------------------------------------------------------------------
__global__ void pack_A_kernel(const float* __restrict__ x,
                              const float* __restrict__ se,
                              const float* __restrict__ si,
                              f16* __restrict__ A) {
    long t   = (long)blockIdx.x * blockDim.x + threadIdx.x;
    long idx = t * 4;
    if (idx >= (long)B_DIM * KTOT) return;
    int b = (int)(idx / KTOT);
    int k = (int)(idx % KTOT);
    v4f val;
    if (k < DIN) {
        val = *(const v4f*)(x + (long)b * DIN + k);
    } else if (k < DIN + NE) {
        val = *(const v4f*)(se + (long)b * NE + (k - DIN));
    } else {
        v4f s = *(const v4f*)(si + (long)b * NI + (k - DIN - NE));
        val = -s;
    }
    v4h h;
#pragma unroll
    for (int e = 0; e < 4; ++e) h[e] = (f16)val[e];
    *(v4h*)(A + idx) = h;
}

// ---------------------------------------------------------------------------
// Pack W = [[W_e_in, relu(W_ee), relu(W_ei)],
//           [W_i_in, relu(W_ie), relu(W_ii)]] as f16, row-major [NTOT, KTOT]
// ---------------------------------------------------------------------------
__global__ void pack_W_kernel(const float* __restrict__ Wein,
                              const float* __restrict__ Wee,
                              const float* __restrict__ Wei,
                              const float* __restrict__ Wiin,
                              const float* __restrict__ Wie,
                              const float* __restrict__ Wii,
                              f16* __restrict__ W) {
    long t   = (long)blockIdx.x * blockDim.x + threadIdx.x;
    long idx = t * 4;
    if (idx >= (long)NTOT * KTOT) return;
    int n = (int)(idx / KTOT);
    int k = (int)(idx % KTOT);
    v4f val;
    if (n < NE) {
        if (k < DIN)           val = *(const v4f*)(Wein + (long)n * DIN + k);
        else if (k < DIN + NE) val = *(const v4f*)(Wee  + (long)n * NE  + (k - DIN));
        else                   val = *(const v4f*)(Wei  + (long)n * NI  + (k - DIN - NE));
    } else {
        int n2 = n - NE;
        if (k < DIN)           val = *(const v4f*)(Wiin + (long)n2 * DIN + k);
        else if (k < DIN + NE) val = *(const v4f*)(Wie  + (long)n2 * NE  + (k - DIN));
        else                   val = *(const v4f*)(Wii  + (long)n2 * NI  + (k - DIN - NE));
    }
    const bool doRelu = (k >= DIN);   // Dale's-law rectification on recurrent weights only
    v4h h;
#pragma unroll
    for (int e = 0; e < 4; ++e) {
        float f = val[e];
        if (doRelu) f = fmaxf(f, 0.0f);
        h[e] = (f16)f;
    }
    *(v4h*)(W + idx) = h;
}

// ---------------------------------------------------------------------------
// Issue one K-stage (32 wide) of async copies into LDS:
//   A tile 128 x 32 (2 x b128 per thread), B tile 256 x 32 (4 x b128 per thread)
//   => 6 ASYNCcnt-tracked ops per thread per stage.
// ---------------------------------------------------------------------------
__device__ __forceinline__ void issue_stage(const f16* __restrict__ A,
                                            const f16* __restrict__ W,
                                            int kt, int m_blk, int n_blk,
                                            f16* ldsA, f16* ldsB, int tid) {
#pragma unroll
    for (int r = 0; r < 2; ++r) {
        const int t   = tid + r * 256;   // 0..511
        const int row = t >> 2;          // 0..127
        const int seg = t & 3;           // 0..3 (8 f16 each)
        cp16_g2lds(ldsA + row * LDA + seg * 8,
                   A + (long)(m_blk + row) * KTOT + kt + seg * 8);
    }
#pragma unroll
    for (int r = 0; r < 4; ++r) {
        const int t   = tid + r * 256;   // 0..1023
        const int row = t >> 2;          // 0..255
        const int seg = t & 3;
        cp16_g2lds(ldsB + row * LDB + seg * 8,
                   W + (long)(n_blk + row) * KTOT + kt + seg * 8);
    }
}

// ---------------------------------------------------------------------------
// WMMA GEMM  I = A @ W^T  [2048 x 5120], LDS double-buffered via async copies,
// fused LIF epilogue.
// Block: 256 threads = 8 waves, 2 (M) x 4 (N); wave tile 64x64.
// Block tile: 128 (M) x 256 (N). Grid: (5120/256, 2048/128) = (20, 16).
// ---------------------------------------------------------------------------
__global__ __launch_bounds__(256) void lif_gemm_kernel(
        const f16* __restrict__ A,     // [B_DIM][KTOT]
        const f16* __restrict__ W,     // [NTOT][KTOT]
        const float* __restrict__ v_e, const float* __restrict__ i_e,
        const float* __restrict__ v_i, const float* __restrict__ i_i,
        float* __restrict__ out) {
    __shared__ __align__(16) f16 sA[2][128 * LDA];   // 2 x 10 KB
    __shared__ __align__(16) f16 sB[2][256 * LDB];   // 2 x 20 KB

    const int tid   = threadIdx.x;
    const int lane  = tid & 31;
    const int wave  = tid >> 5;
    const int wm    = wave >> 2;              // 0..1
    const int wn    = wave & 3;               // 0..3
    const int m_blk = blockIdx.y * 128;
    const int n_blk = blockIdx.x * 256;
    const int lmod  = lane & 15;
    const int lhalf = lane >> 4;

    v8f acc[4][4];
    const v8f vzero = {};
#pragma unroll
    for (int i = 0; i < 4; ++i)
#pragma unroll
        for (int j = 0; j < 4; ++j) acc[i][j] = vzero;

    issue_stage(A, W, 0, m_blk, n_blk, sA[0], sB[0], tid);

#pragma unroll 1
    for (int s = 0; s < NSTAGES; ++s) {
        const int buf = s & 1;
        if (s + 1 < NSTAGES) {
            issue_stage(A, W, (s + 1) * 32, m_blk, n_blk,
                        sA[buf ^ 1], sB[buf ^ 1], tid);
            ASYNC_WAIT(6);   // stage s's 6 ops complete; stage s+1 in flight
        } else {
            ASYNC_WAIT(0);
        }
        __syncthreads();     // whole tile (all waves' portions) resident

        const f16* pA = sA[buf];
        const f16* pB = sB[buf];
        v16h a[4];
#pragma unroll
        for (int im = 0; im < 4; ++im) {
            const f16* p = pA + (wm * 64 + im * 16 + lmod) * LDA + 8 * lhalf;
            union { v16h v; v8h h[2]; } u;
            u.h[0] = *(const v8h*)p;          // K = 8*lhalf + [0..7]
            u.h[1] = *(const v8h*)(p + 16);   // K = 16 + 8*lhalf + [0..7]
            a[im] = u.v;
        }
        v16h b[4];
#pragma unroll
        for (int in = 0; in < 4; ++in)
        {
            const f16* q = pB + (wn * 64 + in * 16 + lmod) * LDB + 8 * lhalf;
            union { v16h v; v8h h[2]; } ub;
            ub.h[0] = *(const v8h*)q;          // K = 8*lhalf + [0..7]
            ub.h[1] = *(const v8h*)(q + 16);   // K = 16 + 8*lhalf + [0..7]
            b[in] = ub.v;
        }
#pragma unroll
        for (int im = 0; im < 4; ++im)
#pragma unroll
            for (int in = 0; in < 4; ++in)
                acc[im][in] = __builtin_amdgcn_wmma_f32_16x16x32_f16(
                    false, a[im], false, b[in], (short)0, acc[im][in],
                    false, false);

        __syncthreads();     // safe to overwrite buffer `buf` next stage
    }

    // Fused LIF epilogue. N-tiles of 256 never straddle col 4096 => uniform.
    const bool isE = (n_blk < NE);
#pragma unroll
    for (int in = 0; in < 4; ++in) {
        const int col = n_blk + wn * 64 + in * 16 + lmod;
#pragma unroll
        for (int im = 0; im < 4; ++im) {
#pragma unroll
            for (int r = 0; r < 8; ++r) {
                const int   row = m_blk + wm * 64 + im * 16 + r + 8 * lhalf;
                const float I   = acc[im][in][r];
                if (isE) {
                    const long  idx   = (long)row * NE + col;
                    const float v     = v_e[idx];
                    const float ic    = i_e[idx];
                    const float v_dec = v + 0.05f * (ic - v);   // tau_mem_e_inv = 1/20
                    const float i_dec = ic * 0.8f;              // 1 - dt/tau_syn
                    const float z     = (v_dec > 1.0f) ? 1.0f : 0.0f;
                    out[OFF_SE + idx] = z;
                    out[OFF_VE + idx] = (1.0f - z) * v_dec;     // V_RESET = 0
                    out[OFF_IE + idx] = i_dec + I;
                } else {
                    const long  idx   = (long)row * NI + (col - NE);
                    const float v     = v_i[idx];
                    const float ic    = i_i[idx];
                    const float v_dec = v + 0.1f * (ic - v);    // tau_mem_i_inv = 1/10
                    const float i_dec = ic * 0.8f;
                    const float z     = (v_dec > 1.0f) ? 1.0f : 0.0f;
                    out[OFF_SI + idx] = z;
                    out[OFF_VI + idx] = (1.0f - z) * v_dec;
                    out[OFF_II + idx] = i_dec + I;
                }
            }
        }
    }
}

extern "C" void kernel_launch(void* const* d_in, const int* in_sizes, int n_in,
                              void* d_out, int out_size, void* d_ws, size_t ws_size,
                              hipStream_t stream) {
    (void)in_sizes; (void)n_in; (void)out_size; (void)ws_size;
    const float* x    = (const float*)d_in[0];
    const float* v_e  = (const float*)d_in[1];
    const float* i_e  = (const float*)d_in[2];
    const float* v_i  = (const float*)d_in[3];
    const float* i_i  = (const float*)d_in[4];
    const float* se   = (const float*)d_in[5];
    const float* si   = (const float*)d_in[6];
    const float* Wee  = (const float*)d_in[7];
    const float* Wie  = (const float*)d_in[8];
    const float* Wei  = (const float*)d_in[9];
    const float* Wii  = (const float*)d_in[10];
    const float* Wein = (const float*)d_in[11];
    const float* Wiin = (const float*)d_in[12];
    float* out = (float*)d_out;

    // Workspace: A_f16 (25 165 824 B, 256-aligned) then W_f16 (62 914 560 B)
    f16* Af = (f16*)d_ws;
    f16* Wf = (f16*)((char*)d_ws + (size_t)B_DIM * KTOT * sizeof(f16));

    {
        long nThreads = (long)B_DIM * KTOT / 4;
        pack_A_kernel<<<(unsigned)((nThreads + 255) / 256), 256, 0, stream>>>(x, se, si, Af);
    }
    {
        long nThreads = (long)NTOT * KTOT / 4;
        pack_W_kernel<<<(unsigned)((nThreads + 255) / 256), 256, 0, stream>>>(
            Wein, Wee, Wei, Wiin, Wie, Wii, Wf);
    }
    {
        dim3 grid(NTOT / 256, B_DIM / 128);
        lif_gemm_kernel<<<grid, 256, 0, stream>>>(Af, Wf, v_e, i_e, v_i, i_i, out);
    }
}
